// FullModelRNN_31301721653842
// MI455X (gfx1250) — matrix operand order repaired and checked by reading, not compiled, not run on hardware
//
#include <hip/hip_runtime.h>
#include <hip/hip_bf16.h>

typedef __attribute__((ext_vector_type(16))) __bf16 v16bf;
typedef __attribute__((ext_vector_type(8)))  __bf16 v8bf;
typedef __attribute__((ext_vector_type(8)))  float  v8f;
typedef __attribute__((__vector_size__(16))) int    vi4;

#define BATCH 2048
#define KSTEPS 256
#define HID 128
#define G3 384
#define LIFT 32

#define AS1 __attribute__((address_space(1)))
#define AS3 __attribute__((address_space(3)))

#if __has_builtin(__builtin_amdgcn_global_load_async_to_lds_b128) && \
    __has_builtin(__builtin_amdgcn_global_load_async_to_lds_b32) &&  \
    __has_builtin(__builtin_amdgcn_s_wait_asynccnt)
#define HAVE_ASYNC 1
#else
#define HAVE_ASYNC 0
#endif

__device__ __forceinline__ float sigf(float x) { return 1.0f / (1.0f + __expf(-x)); }

#if HAVE_ASYNC
__device__ __forceinline__ void async_ld_b128(const float* g, float* l) {
  __builtin_amdgcn_global_load_async_to_lds_b128(
      (AS1 vi4*)(vi4*)(void*)(float*)g,
      (AS3 vi4*)(vi4*)(void*)l, 0, 0);
}
__device__ __forceinline__ void async_ld_b32(const float* g, float* l) {
  __builtin_amdgcn_global_load_async_to_lds_b32(
      (AS1 int*)(int*)(void*)(float*)g,
      (AS3 int*)(int*)(void*)l, 0, 0);
}
#endif

// A-matrix (16x32 bf16) fragment: lane L -> M=L%16, hi=L/16.
// Per ISA: VGPR0..3 hold K = hi*8 + 0..7, VGPR4..7 hold K = 16 + hi*8 + 0..7.
__device__ __forceinline__ v16bf ld_fragA(const __bf16* rowbase, int k0, int hi) {
  const __bf16* p = rowbase + k0 + hi * 8;
  v8bf a = *(const v8bf*)(p);
  v8bf b = *(const v8bf*)(p + 16);
  v16bf r;
#pragma unroll
  for (int i = 0; i < 8; ++i) { r[i] = a[i]; r[i + 8] = b[i]; }
  return r;
}

// B-matrix (32x16 bf16) fragment: lane L -> N=L%16, hi=L/16.
// VGPR0..3 hold K = hi*8 + 0..7, VGPR4..7 hold K = 16 + hi*8 + 0..7 (same as A).
__device__ __forceinline__ v16bf ld_fragB(const __bf16* rowbase, int k0, int hi) {
  const __bf16* p = rowbase + k0 + hi * 8;
  v8bf a = *(const v8bf*)(p);
  v8bf b = *(const v8bf*)(p + 16);
  v16bf r;
#pragma unroll
  for (int i = 0; i < 8; ++i) { r[i] = a[i]; r[i + 8] = b[i]; }
  return r;
}

__device__ __forceinline__ void chain_rhs(const float* y, const float* k, float* d) {
  d[0]  = -k[0] * y[0] + k[12] * y[1];
  d[1]  =  k[0] * y[0] - k[12] * y[1] - k[1] * y[1];
  d[2]  =  k[1] * y[1] - k[2]  * y[2] + k[13] * y[3];
  d[3]  =  k[2] * y[2] - k[13] * y[3] - k[3]  * y[3];
  d[4]  =  k[3] * y[3] - k[4]  * y[4] + k[14] * y[5];
  d[5]  =  k[4] * y[4] - k[14] * y[5] - k[5]  * y[5];
  d[6]  =  k[5] * y[5] - k[6]  * y[6] + k[15] * y[7];
  d[7]  =  k[6] * y[6] - k[15] * y[7] - k[7]  * y[7];
  d[8]  =  k[7] * y[7] - k[8]  * y[8] + k[16] * y[9];
  d[9]  =  k[8] * y[8] - k[16] * y[9] - k[9]  * y[9];
  d[10] =  k[9] * y[9] - k[10] * y[10] + k[17] * y[11];
  d[11] =  k[10] * y[10] - k[17] * y[11] - k[11] * y[11] + k[18] * y[12];
  d[12] =  k[11] * y[11] - k[18] * y[12];
}

__global__ __launch_bounds__(256, 1) void fullmodel_gru_ode_kernel(
    const float* __restrict__ y0, const float* __restrict__ u_seq,
    const float* __restrict__ dt_seq, const float* __restrict__ y_seq,
    const float* __restrict__ W_lift, const float* __restrict__ b_lift,
    const float* __restrict__ w_ih, const float* __restrict__ w_hh,
    const float* __restrict__ b_ih, const float* __restrict__ b_hh,
    const float* __restrict__ W_head, const float* __restrict__ b_head,
    const float* __restrict__ ujump, const int* __restrict__ tf_every_p,
    float* __restrict__ y_out, float* __restrict__ theta_out) {
  __shared__ __align__(16) __bf16 sWhh[G3 * HID];     // [N=384][K=128]
  __shared__ __align__(16) __bf16 sWih[G3 * LIFT];    // [N=384][K=32]
  __shared__ __align__(16) __bf16 sWhd[32 * HID];     // [N=32 pad][K=128]
  __shared__ __align__(16) float  sG[16 * 512];       // rz(0..255), nx(256..383), nh(384..511)
  __shared__ __align__(16) float  sH[16 * HID];       // fp32 hidden state
  __shared__ __align__(16) __bf16 sHb[16 * HID];      // bf16 hidden (A operand)
  __shared__ __align__(16) __bf16 sLift[16 * LIFT];   // bf16 lifted (A operand)
  __shared__ __align__(16) float  sWlift[9 * 32];
  __shared__ __align__(16) float  sUst[2 * 16 * 4];   // async double-buffered u rows
  __shared__ __align__(16) float  sDst[2 * 16];       // async double-buffered dt
  __shared__ float sBlift[32];
  __shared__ float sBih[G3];
  __shared__ float sBhh[G3];
  __shared__ float sBhead[19];
  __shared__ float sJump[20];
  __shared__ float sTheta[16 * 20];
  __shared__ float sY[16 * 16];                       // 13 species, padded stride 16
  __shared__ float sFeat[16 * 12];                    // 9 feats, padded stride 12

  const int tid = threadIdx.x;
  const int lane = tid & 31;
  const int wid = tid >> 5;
  const int hi = lane >> 4;
  const int nl = lane & 15;
  const int b0 = blockIdx.x * 16;
  const int tf_every = tf_every_p[0];

  // ---- one-time: weights -> LDS (bf16) ----
  for (int i = tid; i < G3 * HID; i += 256) sWhh[i] = (__bf16)w_hh[i];
  for (int i = tid; i < G3 * LIFT; i += 256) sWih[i] = (__bf16)w_ih[i];
  for (int i = tid; i < 32 * HID; i += 256) {
    int n = i >> 7, kk = i & 127;
    sWhd[i] = (n < 19) ? (__bf16)W_head[kk * 19 + n] : (__bf16)0.0f;
  }
  for (int i = tid; i < 9 * 32; i += 256) sWlift[i] = W_lift[i];
  if (tid < 32) sBlift[tid] = b_lift[tid];
  for (int i = tid; i < G3; i += 256) { sBih[i] = b_ih[i]; sBhh[i] = b_hh[i]; }
  if (tid < 19) sBhead[tid] = b_head[tid];
  if (tid < 20) sJump[tid] = ujump[tid];
  for (int i = tid; i < 16 * HID; i += 256) { sH[i] = 0.0f; sHb[i] = (__bf16)0.0f; }
  for (int i = tid; i < 16 * 13; i += 256) {
    int m = i / 13, s = i - 13 * m;
    float v = 0.01f;
    if (s % 3 == 0) v = y0[(size_t)(b0 + m) * 5 + s / 3] + 0.01f;  // OBS = 0,3,6,9,12
    sY[m * 16 + s] = v;
  }

#if HAVE_ASYNC
  // Pre-issue async staging of u/dt rows for k=0 into buffer 0.
  if (wid == 0) {
    if (lane < 16) {
      async_ld_b128(u_seq + ((size_t)(b0 + lane) * KSTEPS + 0) * 4, sUst + lane * 4);
    } else {
      int m = lane - 16;
      async_ld_b32(dt_seq + (size_t)(b0 + m) * KSTEPS + 0, sDst + m);
    }
  }
#endif
  __syncthreads();

  for (int k = 0; k < KSTEPS; ++k) {
#if HAVE_ASYNC
    // wait for this step's staged u/dt, make visible, then issue next step's
    if (wid == 0) __builtin_amdgcn_s_wait_asynccnt(0);
    __syncthreads();
    if (wid == 0 && (k + 1) < KSTEPS) {
      const int nb = (k + 1) & 1;
      if (lane < 16) {
        async_ld_b128(u_seq + ((size_t)(b0 + lane) * KSTEPS + (k + 1)) * 4,
                      sUst + nb * 64 + lane * 4);
      } else {
        int m = lane - 16;
        async_ld_b32(dt_seq + (size_t)(b0 + m) * KSTEPS + (k + 1), sDst + nb * 16 + m);
      }
    }
#endif

    // ---- feat = [u_k, y_prev_in] ----
    if (tid < 144) {
      int m = tid / 9, j = tid - 9 * m;
      float f;
      if (j < 4) {
#if HAVE_ASYNC
        f = sUst[(k & 1) * 64 + m * 4 + j];
#else
        f = u_seq[((size_t)(b0 + m) * KSTEPS + k) * 4 + j];
#endif
      } else {
        int p = j - 4;
        bool tf = (k > 0) && ((k % tf_every) == 0);
        f = tf ? y_seq[((size_t)(b0 + m) * KSTEPS + (k - 1)) * 5 + p] : sY[m * 16 + p * 3];
      }
      sFeat[m * 12 + j] = f;
    }
    if (tid < 16 && (k + 1) < KSTEPS && (((k + 1) % tf_every) == 0))
      __builtin_prefetch(&y_seq[((size_t)(b0 + tid) * KSTEPS + k) * 5], 0, 1);
    __syncthreads();

    // ---- lift: silu(feat @ W_lift + b_lift), 16x32 ----
#pragma unroll
    for (int rr = 0; rr < 2; ++rr) {
      int i = tid + rr * 256;
      int m = i >> 5, j = i & 31;
      float acc = sBlift[j];
#pragma unroll
      for (int t2 = 0; t2 < 9; ++t2) acc += sFeat[m * 12 + t2] * sWlift[t2 * 32 + j];
      acc = acc * sigf(acc);
      sLift[m * 32 + j] = (__bf16)acc;
    }
    __syncthreads();

    // ---- gate GEMMs via WMMA: gx (K=32) + gh (K=128); n-gate kept split ----
    {
      v16bf aL = ld_fragA(sLift + nl * 32, 0, hi);
      v16bf aH[4];
#pragma unroll
      for (int kk = 0; kk < 4; ++kk) aH[kk] = ld_fragA(sHb + nl * HID, kk * 32, hi);
#pragma unroll
      for (int tt = 0; tt < 3; ++tt) {
        // scalarize the (wave-uniform) tile id so the r/z-vs-n branch is an SALU
        // branch and EXEC stays all-ones across WMMA.
        const int t = __builtin_amdgcn_readfirstlane(wid * 3 + tt);  // 0..23
        const int ncol = t * 16 + nl;
        v16bf bI = ld_fragB(sWih + ncol * 32, 0, hi);
        if (t < 16) {  // r/z gates: fuse gx+gh+biases
          const float bias = sBih[ncol] + sBhh[ncol];
          v8f c;
#pragma unroll
          for (int v = 0; v < 8; ++v) c[v] = bias;
          c = __builtin_amdgcn_wmma_f32_16x16x32_bf16(false, aL, false, bI, (short)0, c, false, false);
#pragma unroll
          for (int kk = 0; kk < 4; ++kk) {
            v16bf bH = ld_fragB(sWhh + ncol * HID, kk * 32, hi);
            c = __builtin_amdgcn_wmma_f32_16x16x32_bf16(false, aH[kk], false, bH, (short)0, c, false, false);
          }
#pragma unroll
          for (int v = 0; v < 8; ++v) sG[(v + hi * 8) * 512 + ncol] = c[v];
        } else {  // n gate: nx and nh kept separate
          v8f c;
#pragma unroll
          for (int v = 0; v < 8; ++v) c[v] = sBih[ncol];
          c = __builtin_amdgcn_wmma_f32_16x16x32_bf16(false, aL, false, bI, (short)0, c, false, false);
#pragma unroll
          for (int v = 0; v < 8; ++v) sG[(v + hi * 8) * 512 + ncol] = c[v];
          v8f c2;
#pragma unroll
          for (int v = 0; v < 8; ++v) c2[v] = sBhh[ncol];
#pragma unroll
          for (int kk = 0; kk < 4; ++kk) {
            v16bf bH = ld_fragB(sWhh + ncol * HID, kk * 32, hi);
            c2 = __builtin_amdgcn_wmma_f32_16x16x32_bf16(false, aH[kk], false, bH, (short)0, c2, false, false);
          }
#pragma unroll
          for (int v = 0; v < 8; ++v) sG[(v + hi * 8) * 512 + (ncol + 128)] = c2[v];
        }
      }
    }
    __syncthreads();

    // ---- GRU elementwise update ----
#pragma unroll
    for (int rr = 0; rr < 8; ++rr) {
      int i = tid + rr * 256;
      int m = i >> 7, j = i & 127;
      const float* gm = sG + m * 512;
      float r = sigf(gm[j]);
      float z = sigf(gm[128 + j]);
      float n = tanhf(gm[256 + j] + r * gm[384 + j]);
      float hn = (1.0f - z) * n + z * sH[i];
      sH[i] = hn;
      sHb[i] = (__bf16)hn;
    }
    __syncthreads();

    // ---- head: theta = gamma(h_new @ W_head + b_head), N=19 (2 tiles) ----
    if (__builtin_amdgcn_readfirstlane(wid) < 2) {
      const int ncol = wid * 16 + nl;
      float bias = (ncol < 19) ? sBhead[ncol] : 0.0f;
      v8f c;
#pragma unroll
      for (int v = 0; v < 8; ++v) c[v] = bias;
#pragma unroll
      for (int kk = 0; kk < 4; ++kk) {
        v16bf aH = ld_fragA(sHb + nl * HID, kk * 32, hi);
        v16bf bH = ld_fragB(sWhd + ncol * HID, kk * 32, hi);
        c = __builtin_amdgcn_wmma_f32_16x16x32_bf16(false, aH, false, bH, (short)0, c, false, false);
      }
#pragma unroll
      for (int v = 0; v < 8; ++v) {
        int m = v + hi * 8;
        if (ncol < 19) {
          float th = 0.001f + 1.999f * sigf(c[v]);
          sTheta[m * 20 + ncol] = th;
          theta_out[((size_t)(b0 + m) * KSTEPS + k) * 19 + ncol] = th;
        }
      }
    }
    __syncthreads();

    // ---- jump + RK4, one thread per batch row ----
    if (tid < 16) {
      const int m = tid, bg = b0 + m;
      float y[13], th[19], d1[13], d2[13], d3[13], d4[13], yt[13];
#pragma unroll
      for (int s = 0; s < 13; ++s) y[s] = sY[m * 16 + s];
#pragma unroll
      for (int j = 0; j < 19; ++j) th[j] = sTheta[m * 20 + j];
      float u0 = sFeat[m * 12 + 0], u1 = sFeat[m * 12 + 1];
      float u2 = sFeat[m * 12 + 2], u3 = sFeat[m * 12 + 3];
#pragma unroll
      for (int p = 0; p < 5; ++p)
        y[p * 3] += u0 * sJump[p] + u1 * sJump[5 + p] + u2 * sJump[10 + p] + u3 * sJump[15 + p];
#if HAVE_ASYNC
      const float dt = sDst[(k & 1) * 16 + m];
#else
      const float dt = dt_seq[(size_t)bg * KSTEPS + k];
#endif
      chain_rhs(y, th, d1);
#pragma unroll
      for (int s = 0; s < 13; ++s) yt[s] = fmaf(0.5f * dt, d1[s], y[s]);
      chain_rhs(yt, th, d2);
#pragma unroll
      for (int s = 0; s < 13; ++s) yt[s] = fmaf(0.5f * dt, d2[s], y[s]);
      chain_rhs(yt, th, d3);
#pragma unroll
      for (int s = 0; s < 13; ++s) yt[s] = fmaf(dt, d3[s], y[s]);
      chain_rhs(yt, th, d4);
#pragma unroll
      for (int s = 0; s < 13; ++s) {
        float yn = y[s] + dt * (1.0f / 6.0f) * (d1[s] + 2.0f * d2[s] + 2.0f * d3[s] + d4[s]);
        y[s] = fmaxf(yn, 0.0f);
        sY[m * 16 + s] = y[s];
      }
#pragma unroll
      for (int p = 0; p < 5; ++p)
        y_out[((size_t)bg * KSTEPS + k) * 5 + p] = y[p * 3];
    }
    __syncthreads();
  }
}

extern "C" void kernel_launch(void* const* d_in, const int* in_sizes, int n_in,
                              void* d_out, int out_size, void* d_ws, size_t ws_size,
                              hipStream_t stream) {
  const float* y0     = (const float*)d_in[0];
  const float* u_seq  = (const float*)d_in[1];
  const float* dt_seq = (const float*)d_in[2];
  const float* y_seq  = (const float*)d_in[3];
  const float* W_lift = (const float*)d_in[4];
  const float* b_lift = (const float*)d_in[5];
  const float* w_ih   = (const float*)d_in[6];
  const float* w_hh   = (const float*)d_in[7];
  const float* b_ih   = (const float*)d_in[8];
  const float* b_hh   = (const float*)d_in[9];
  const float* W_head = (const float*)d_in[10];
  const float* b_head = (const float*)d_in[11];
  const float* ujump  = (const float*)d_in[12];
  const int*   tf_ev  = (const int*)d_in[13];

  float* y_out = (float*)d_out;
  float* theta_out = y_out + (size_t)BATCH * KSTEPS * 5;

  fullmodel_gru_ode_kernel<<<BATCH / 16, 256, 0, stream>>>(
      y0, u_seq, dt_seq, y_seq, W_lift, b_lift, w_ih, w_hh, b_ih, b_hh,
      W_head, b_head, ujump, tf_ev, y_out, theta_out);
}
